// Attention_16965120820033
// MI455X (gfx1250) — matrix operand order repaired and checked by reading, not compiled, not run on hardware
//
#include <hip/hip_runtime.h>
#include <hip/hip_bf16.h>
#include <stdint.h>

// ---------------------------------------------------------------------------
// CDNA5 (gfx1250) fused Bayesian multi-head attention.
// wave32, WMMA bf16 16x16x32 (f32 accum), TDM global->LDS staging with D# pad.
// All B operands kept N-major so every fragment is a contiguous per-lane load.
// ---------------------------------------------------------------------------

typedef __bf16 bf16;
typedef __attribute__((ext_vector_type(16))) __bf16 v16bf;
typedef __attribute__((ext_vector_type(8)))  __bf16 v8bf;
typedef __attribute__((ext_vector_type(8)))  float  v8f;
typedef __attribute__((ext_vector_type(4)))  unsigned int v4u;
typedef __attribute__((ext_vector_type(8)))  int v8i;
typedef __attribute__((ext_vector_type(4)))  int v4i;

#define C_DIM 1024
#define HEADS 16
#define HDIM  64
#define NSEQ  2048
#define BM    8192   // B*N rows

#if __has_builtin(__builtin_amdgcn_tensor_load_to_lds)
#define HAVE_TDM 1
#else
#define HAVE_TDM 0
#endif

// ------------------------------ WMMA helpers -------------------------------

__device__ __forceinline__ v8f wmma_bf16(v16bf a, v16bf b, v8f c) {
  return __builtin_amdgcn_wmma_f32_16x16x32_bf16(
      /*neg_a=*/false, a, /*neg_b=*/false, b,
      /*c_mod=*/(short)0, c, /*reuse_a=*/false, /*reuse_b=*/false);
}

// A-matrix 16x32 bf16 fragment from row-major tile (lda multiple of 8).
// lanes 0-15: K=0-15 ; lanes 16-31: K=16-31.
// Two aligned 16-byte loads per lane.
__device__ __forceinline__ v16bf load_a_frag(const bf16* __restrict__ base,
                                             int lda, int lane) {
  const int row = lane & 15;
  const int ko  = (lane < 16) ? 0 : 16;
  const bf16* p = base + row * lda + ko;
  v8bf lo = *(const v8bf*)(p);        // K = ko .. ko+7
  v8bf hi = *(const v8bf*)(p + 8);    // K = ko+8 .. ko+15
  return __builtin_shufflevector(lo, hi, 0, 1, 2, 3, 4, 5, 6, 7,
                                 8, 9, 10, 11, 12, 13, 14, 15);
}

// B-matrix 32x16 bf16 fragment from an N-major tile (ldb multiple of 16):
// element (K, col) at base[col*ldb + K].  lanes 0-15: col, K=0-15;
// lanes 16-31: col, K=16-31.  One aligned 32-byte load per lane.
__device__ __forceinline__ v16bf load_b_frag(const bf16* __restrict__ base,
                                             int ldb, int lane) {
  const int col = lane & 15;
  const int kb  = (lane < 16) ? 0 : 16;
  return *(const v16bf*)(base + col * ldb + kb);
}

__device__ __forceinline__ float softplusf(float r) {
  return fmaxf(r, 0.0f) + log1pf(__expf(-fabsf(r)));
}

// ------------------------ Tensor Data Mover (TDM) --------------------------
#if HAVE_TDM
// 2D tile of bf16 (data_size=2B) global -> LDS, D# per ISA 8.3/8.4.
// pad_iv: 0=2,1=4,2=8,3=16,4=32... DWORDs between pads; pad_amt: 0=1..7=8 DWORDs.
__device__ __forceinline__ void tdm_load_2d_bf16(
    unsigned lds_off, const void* gptr,
    unsigned tensor_w, unsigned tensor_h, unsigned row_stride,
    unsigned tile_w, unsigned tile_h,
    unsigned pad_en, unsigned pad_iv, unsigned pad_amt) {
  unsigned long long ga = (unsigned long long)gptr;
  v4u g0;
  g0.x = 1u;                                      // count=1, user descriptor
  g0.y = lds_off;                                 // lds_addr
  g0.z = (unsigned)(ga & 0xffffffffu);            // global_addr lo
  g0.w = (unsigned)((ga >> 32) & 0x01ffffffu) | (2u << 30);  // addr hi | type=2
  v8i g1;
  g1[0] = (int)((1u << 16) |                      // data_size=1 -> 2 bytes
                (pad_en << 20) | (pad_iv << 22) | (pad_amt << 25));
  g1[1] = (int)((tensor_w & 0xffffu) << 16);      // tensor_dim0[15:0]
  g1[2] = (int)(((tensor_w >> 16) & 0xffffu) |    // tensor_dim0[31:16]
                ((tensor_h & 0xffffu) << 16));    // tensor_dim1[15:0]
  g1[3] = (int)(((tensor_h >> 16) & 0xffffu) |    // tensor_dim1[31:16]
                ((tile_w & 0xffffu) << 16));      // tile_dim0
  g1[4] = (int)(tile_h & 0xffffu);                // tile_dim1 (tile_dim2=0)
  g1[5] = (int)row_stride;                        // tensor_dim0_stride lo
  g1[6] = 0;
  g1[7] = 0;
  v4i z4 = {0, 0, 0, 0};
  v8i z8 = {0, 0, 0, 0, 0, 0, 0, 0};
  __builtin_amdgcn_tensor_load_to_lds(g0, g1, z4, z4, z8, 0);
}
#endif

// ------------------------------ Prep kernels -------------------------------
// No transposes needed anywhere: weights stay N-major ([n][k]) for WMMA B.

__global__ __launch_bounds__(256) void k_cvt_bf16(
    const float* __restrict__ in, bf16* __restrict__ out, int n) {
  const int idx = (blockIdx.x * 256 + threadIdx.x) * 4;
  if (idx + 3 < n) {
    float4 v = *(const float4*)(in + idx);
    out[idx + 0] = (bf16)v.x;
    out[idx + 1] = (bf16)v.y;
    out[idx + 2] = (bf16)v.z;
    out[idx + 3] = (bf16)v.w;
  }
}

__global__ __launch_bounds__(256) void k_bayes_bf16(
    const float* __restrict__ mu, const float* __restrict__ rho,
    const float* __restrict__ eps, bf16* __restrict__ out, int n) {
  const int idx = (blockIdx.x * 256 + threadIdx.x) * 4;
  if (idx + 3 < n) {
    float4 m = *(const float4*)(mu + idx);
    float4 r = *(const float4*)(rho + idx);
    float4 e = *(const float4*)(eps + idx);
    out[idx + 0] = (bf16)(m.x + softplusf(r.x) * e.x);
    out[idx + 1] = (bf16)(m.y + softplusf(r.y) * e.y);
    out[idx + 2] = (bf16)(m.z + softplusf(r.z) * e.z);
    out[idx + 3] = (bf16)(m.w + softplusf(r.w) * e.w);
  }
}

__global__ __launch_bounds__(256) void k_prep_bias(
    const float* __restrict__ mu, const float* __restrict__ rho,
    const float* __restrict__ eps, float* __restrict__ out) {
  const int i = blockIdx.x * 256 + threadIdx.x;
  if (i < C_DIM) out[i] = mu[i] + softplusf(rho[i]) * eps[i];
}

// ------------------------------- GEMM kernel -------------------------------
// Y[M=8192, N=1024] = A[M,K] @ W^T where W is [N,K] row-major (N-major B).
// mode 0: bf16 out scatter [B,H,N,D]; mode 1: bf16 out scatter [B,H,D,N];
// mode 2: f32 out [M,N] + bias.
#define GLDA 48   // padded LDS row stride (32 data + 16 pad elems)

__global__ __launch_bounds__(256) void k_gemm_bf16(
    const bf16* __restrict__ A, const bf16* __restrict__ W,
    void* __restrict__ out, const float* __restrict__ bias, int mode) {
  __shared__ bf16 sA[128 * GLDA];   // [128 m rows][32 k] (+pad)
  __shared__ bf16 sB[128 * GLDA];   // [128 n rows][32 k] (+pad)

  const int lane = threadIdx.x & 31;
  const int wave = threadIdx.x >> 5;      // 8 waves
  const int n0 = blockIdx.x * 128;
  const int m0 = blockIdx.y * 128;
  const int wm = wave & 3;                // rows wm*32 .. +32
  const int wn = wave >> 2;               // cols wn*64 .. +64

  v8f acc[2][4];
#pragma unroll
  for (int r = 0; r < 2; ++r)
#pragma unroll
    for (int t = 0; t < 4; ++t)
#pragma unroll
      for (int j = 0; j < 8; ++j) acc[r][t][j] = 0.0f;

  for (int k0 = 0; k0 < C_DIM; k0 += 32) {
    __syncthreads();   // WAR vs previous iteration
#if HAVE_TDM
    if (wave == 0) {
      tdm_load_2d_bf16((unsigned)(unsigned long long)(const void*)sA,
                       A + (size_t)m0 * C_DIM + k0,
                       C_DIM, BM, C_DIM, /*tile_w=*/32, /*tile_h=*/128,
                       /*pad*/1, /*iv: 16 dw*/3, /*amt: 8 dw*/7);
    }
    if (wave == 1) {
      tdm_load_2d_bf16((unsigned)(unsigned long long)(const void*)sB,
                       W + (size_t)n0 * C_DIM + k0,
                       C_DIM, C_DIM, C_DIM, /*tile_w=*/32, /*tile_h=*/128,
                       1, 3, 7);
    }
    __builtin_amdgcn_s_wait_tensorcnt(0);
#else
    for (int e = threadIdx.x; e < 128 * 32; e += 256) {
      const int r = e >> 5, c = e & 31;
      sA[r * GLDA + c] = A[(size_t)(m0 + r) * C_DIM + k0 + c];
      sB[r * GLDA + c] = W[(size_t)(n0 + r) * C_DIM + k0 + c];
    }
#endif
    __syncthreads();

    v16bf af[2], bfr[4];
#pragma unroll
    for (int r = 0; r < 2; ++r)
      af[r] = load_a_frag(sA + (wm * 32 + r * 16) * GLDA, GLDA, lane);
#pragma unroll
    for (int t = 0; t < 4; ++t)
      bfr[t] = load_b_frag(sB + (wn * 64 + t * 16) * GLDA, GLDA, lane);
#pragma unroll
    for (int r = 0; r < 2; ++r)
#pragma unroll
      for (int t = 0; t < 4; ++t)
        acc[r][t] = wmma_bf16(af[r], bfr[t], acc[r][t]);
  }

  // C layout: elem j -> row j + 8*(lane>=16), col lane&15.
  const int hi8 = (lane >> 4) * 8;
  const int cl  = lane & 15;
#pragma unroll
  for (int r = 0; r < 2; ++r) {
#pragma unroll
    for (int t = 0; t < 4; ++t) {
#pragma unroll
      for (int j = 0; j < 8; ++j) {
        const int m = m0 + wm * 32 + r * 16 + j + hi8;
        const int c = n0 + wn * 64 + t * 16 + cl;
        const float v = acc[r][t][j];
        if (mode == 2) {
          ((float*)out)[(size_t)m * C_DIM + c] = v + bias[c];
        } else {
          const int b = m >> 11, n = m & (NSEQ - 1);
          const int h = c >> 6, d = c & (HDIM - 1);
          bf16* ob = (bf16*)out;
          if (mode == 0)   // [B,H,N,D]
            ob[(((size_t)(b * HEADS + h)) * NSEQ + n) * HDIM + d] = (bf16)v;
          else             // [B,H,D,N]
            ob[(((size_t)(b * HEADS + h)) * HDIM + d) * NSEQ + n] = (bf16)v;
        }
      }
    }
  }
}

// ---------------------------- Flash attention ------------------------------
// grid = (N/128 q-blocks, B*H heads), block = 256 (8 waves, 16 q rows each).
// K is [B,H,N,D] (m-major: contiguous d => S B-operand),
// V is [B,H,D,N] (d-major: contiguous m => PV B-operand).
#define ALDK 80   // 64 data + 16 pad elems
#define ALDP 72   // P staging stride

__global__ __launch_bounds__(256) void k_flash_attn(
    const bf16* __restrict__ Q,    // [B,H,N,D]
    const bf16* __restrict__ K,    // [B,H,N,D]
    const bf16* __restrict__ Vt,   // [B,H,D,N]
    bf16* __restrict__ Oc) {       // [B*N, C]
  __shared__ bf16 sK[64 * ALDK];        // [m=64][d=64] (+pad)
  __shared__ bf16 sV[64 * ALDK];        // [d=64][m=64] (+pad)
  __shared__ bf16 sP[8 * 16 * ALDP];    // per-wave 16x64 probs (+pad)

  const int head = blockIdx.y;                   // b*16 + h
  const int b = head >> 4, h = head & 15;
  const int q0 = blockIdx.x * 128;
  const int lane = threadIdx.x & 31;
  const int wave = threadIdx.x >> 5;
  const float scale = 0.125f;                    // 64^-0.5

  const bf16* qbase = Q + ((size_t)head * NSEQ + q0 + wave * 16) * HDIM;
  v16bf qa[2];
  qa[0] = load_a_frag(qbase, HDIM, lane);
  qa[1] = load_a_frag(qbase + 32, HDIM, lane);

  const bf16* kbase = K + (size_t)head * NSEQ * HDIM;   // [n][d]
  const bf16* vbase = Vt + (size_t)head * HDIM * NSEQ;  // [d][n]

  v8f o[4];
  float mrun[8], lrun[8];
#pragma unroll
  for (int t = 0; t < 4; ++t)
#pragma unroll
    for (int j = 0; j < 8; ++j) o[t][j] = 0.0f;
#pragma unroll
  for (int j = 0; j < 8; ++j) { mrun[j] = -1e30f; lrun[j] = 0.0f; }

  const int hi1 = lane >> 4;
  const int cl  = lane & 15;

  for (int m0 = 0; m0 < NSEQ; m0 += 64) {
    __syncthreads();
#if HAVE_TDM
    if (wave == 0) {   // K tile: rows m (contiguous d), 64x64, pad to ALDK
      tdm_load_2d_bf16((unsigned)(unsigned long long)(const void*)sK,
                       kbase + (size_t)m0 * HDIM,
                       HDIM, NSEQ, HDIM, /*tile_w=*/64, /*tile_h=*/64,
                       /*pad*/1, /*iv: 32 dw*/4, /*amt: 8 dw*/7);
    }
    if (wave == 1) {   // V tile: rows d (contiguous m), 64x64, pad to ALDK
      tdm_load_2d_bf16((unsigned)(unsigned long long)(const void*)sV,
                       vbase + m0,
                       NSEQ, HDIM, NSEQ, /*tile_w=*/64, /*tile_h=*/64,
                       1, 4, 7);
    }
    __builtin_amdgcn_s_wait_tensorcnt(0);
#else
    for (int e = threadIdx.x; e < 64 * 64; e += 256) {
      const int r = e >> 6, c = e & 63;
      sK[r * ALDK + c] = kbase[(size_t)(m0 + r) * HDIM + c];
      sV[r * ALDK + c] = vbase[(size_t)r * NSEQ + m0 + c];
    }
#endif
    __syncthreads();

    // S = Q @ K^T : 4 key tiles, K-dim = d (2 fragments of 32)
    v8f s[4];
#pragma unroll
    for (int t = 0; t < 4; ++t) {
#pragma unroll
      for (int j = 0; j < 8; ++j) s[t][j] = 0.0f;
#pragma unroll
      for (int f = 0; f < 2; ++f) {
        v16bf bk = load_b_frag(sK + (t * 16) * ALDK + f * 32, ALDK, lane);
        s[t] = wmma_bf16(qa[f], bk, s[t]);
      }
    }

    // online softmax over this wave's 16 rows
#pragma unroll
    for (int j = 0; j < 8; ++j) {
      float mx = -1e30f;
#pragma unroll
      for (int t = 0; t < 4; ++t) {
        s[t][j] *= scale;
        mx = fmaxf(mx, s[t][j]);
      }
#pragma unroll
      for (int msk = 1; msk < 16; msk <<= 1)
        mx = fmaxf(mx, __shfl_xor(mx, msk, 32));
      const float mn = fmaxf(mrun[j], mx);
      const float alpha = __expf(mrun[j] - mn);
      mrun[j] = mn;
      lrun[j] *= alpha;
#pragma unroll
      for (int t = 0; t < 4; ++t) o[t][j] *= alpha;
      float rs = 0.0f;
#pragma unroll
      for (int t = 0; t < 4; ++t) {
        const float p = __expf(s[t][j] - mn);
        s[t][j] = p;
        rs += p;
      }
#pragma unroll
      for (int msk = 1; msk < 16; msk <<= 1)
        rs += __shfl_xor(rs, msk, 32);
      lrun[j] += rs;
    }

    // stage P (C-layout -> row-major bf16), wave-private region
    bf16* pw = sP + wave * 16 * ALDP;
#pragma unroll
    for (int t = 0; t < 4; ++t)
#pragma unroll
      for (int j = 0; j < 8; ++j)
        pw[(j + 8 * hi1) * ALDP + t * 16 + cl] = (bf16)s[t][j];

    // O += P @ V : K-dim = m (2 fragments of 32), 4 output tiles over d
#pragma unroll
    for (int c = 0; c < 2; ++c) {
      v16bf pa = load_a_frag(pw + c * 32, ALDP, lane);
#pragma unroll
      for (int t = 0; t < 4; ++t) {
        v16bf bv = load_b_frag(sV + (t * 16) * ALDK + c * 32, ALDK, lane);
        o[t] = wmma_bf16(pa, bv, o[t]);
      }
    }
  }

  // normalize, scatter to [B*N, C] bf16 for the projection GEMM
  float inv[8];
#pragma unroll
  for (int j = 0; j < 8; ++j) inv[j] = 1.0f / lrun[j];
#pragma unroll
  for (int t = 0; t < 4; ++t) {
#pragma unroll
    for (int j = 0; j < 8; ++j) {
      const int n = q0 + wave * 16 + j + 8 * hi1;
      const int m = b * NSEQ + n;
      const int c = h * HDIM + t * 16 + cl;
      Oc[(size_t)m * C_DIM + c] = (bf16)(o[t][j] * inv[j]);
    }
  }
}

// ------------------------------- launcher ----------------------------------

extern "C" void kernel_launch(void* const* d_in, const int* in_sizes, int n_in,
                              void* d_out, int out_size, void* d_ws, size_t ws_size,
                              hipStream_t stream) {
  const float* x        = (const float*)d_in[0];
  const float* q_w      = (const float*)d_in[1];
  const float* k_w      = (const float*)d_in[2];
  const float* v_mu     = (const float*)d_in[3];
  const float* v_rho    = (const float*)d_in[4];
  const float* v_eps    = (const float*)d_in[5];
  const float* proj_mu  = (const float*)d_in[6];
  const float* proj_rho = (const float*)d_in[7];
  const float* proj_eps = (const float*)d_in[8];
  const float* pb_mu    = (const float*)d_in[9];
  const float* pb_rho   = (const float*)d_in[10];
  const float* pb_eps   = (const float*)d_in[11];
  float* out = (float*)d_out;

  size_t off = 0;
  char* base = (char*)d_ws;
  auto carve = [&](size_t bytes) -> void* {
    void* p = base + off;
    off += (bytes + 255) & ~(size_t)255;
    return p;
  };
  bf16*  xbf    = (bf16*)carve((size_t)BM * C_DIM * 2);     // x bf16 [M,K]
  bf16*  wq     = (bf16*)carve((size_t)C_DIM * C_DIM * 2);  // q_w bf16 [N,K]
  bf16*  wk     = (bf16*)carve((size_t)C_DIM * C_DIM * 2);
  bf16*  wv     = (bf16*)carve((size_t)C_DIM * C_DIM * 2);  // sampled v_w
  bf16*  wp     = (bf16*)carve((size_t)C_DIM * C_DIM * 2);  // sampled p_w
  float* pbw    = (float*)carve((size_t)C_DIM * 4);         // sampled bias
  bf16*  qws    = (bf16*)carve((size_t)BM * C_DIM * 2);     // [B,H,N,D]
  bf16*  kws    = (bf16*)carve((size_t)BM * C_DIM * 2);     // [B,H,N,D]
  bf16*  vtws   = (bf16*)carve((size_t)BM * C_DIM * 2);     // [B,H,D,N]
  bf16*  attnws = (bf16*)carve((size_t)BM * C_DIM * 2);     // [B*N, C]

  // 1) prep: convert / Bayesian sample (all elementwise, no transposes)
  const int wgrid = (C_DIM * C_DIM) / (256 * 4);
  k_cvt_bf16<<<(BM * C_DIM) / (256 * 4), 256, 0, stream>>>(x, xbf, BM * C_DIM);
  k_cvt_bf16<<<wgrid, 256, 0, stream>>>(q_w, wq, C_DIM * C_DIM);
  k_cvt_bf16<<<wgrid, 256, 0, stream>>>(k_w, wk, C_DIM * C_DIM);
  k_bayes_bf16<<<wgrid, 256, 0, stream>>>(v_mu, v_rho, v_eps, wv, C_DIM * C_DIM);
  k_bayes_bf16<<<wgrid, 256, 0, stream>>>(proj_mu, proj_rho, proj_eps, wp,
                                          C_DIM * C_DIM);
  k_prep_bias<<<C_DIM / 256, 256, 0, stream>>>(pb_mu, pb_rho, pb_eps, pbw);

  // 2) Q/K/V projections (WMMA GEMM, TDM-staged tiles)
  const dim3 gg(C_DIM / 128, BM / 128);
  k_gemm_bf16<<<gg, 256, 0, stream>>>(xbf, wq, qws, nullptr, 0);
  k_gemm_bf16<<<gg, 256, 0, stream>>>(xbf, wk, kws, nullptr, 0);
  k_gemm_bf16<<<gg, 256, 0, stream>>>(xbf, wv, vtws, nullptr, 1);

  // 3) fused flash attention
  k_flash_attn<<<dim3(NSEQ / 128, 4 * HEADS), 256, 0, stream>>>(
      qws, kws, vtws, attnws);

  // 4) output projection + Bayesian bias (f32 result)
  k_gemm_bf16<<<gg, 256, 0, stream>>>(attnws, wp, (void*)out, pbw, 2);
}
